// DifferentialAttention_48473000903241
// MI455X (gfx1250) — matrix operand order repaired and checked by reading, not compiled, not run on hardware
//
#include <hip/hip_runtime.h>

// ---------------------------------------------------------------------------
// Differential Attention for MI455X (gfx1250, wave32, WMMA bf16 16x16x32)
// B=4, T=1024, C=1024, NH=16, HD=64
// GEMMs use double-buffered async global->LDS staging (ASYNCcnt path).
// ---------------------------------------------------------------------------

#define LAMBDA_INIT   0.3555090675909693f
#define ONE_MINUS_LI  0.6444909324090307f

static constexpr int Bn  = 4;
static constexpr int Tn  = 1024;
static constexpr int Cn  = 1024;
static constexpr int NHn = 16;
static constexpr int HDn = 64;

typedef __attribute__((ext_vector_type(16))) __bf16 bf16x16;
typedef __attribute__((ext_vector_type(8)))  __bf16 bf16x8;
typedef __attribute__((ext_vector_type(8)))  float  f32x8;

#define WMMA_BF16(A_, B_, C_) \
  __builtin_amdgcn_wmma_f32_16x16x32_bf16(false, (A_), false, (B_), (short)0, (C_), false, false)

// ---- CDNA5 async global->LDS copy (ASYNCcnt) ------------------------------
__device__ __forceinline__ void async_b128(unsigned lds_off, const __bf16* g) {
  asm volatile("global_load_async_to_lds_b128 %0, %1, off"
               :: "v"(lds_off), "v"(g) : "memory");
}
__device__ __forceinline__ void wait_async0() {
  asm volatile("s_wait_asynccnt 0x0" ::: "memory");
}
__device__ __forceinline__ unsigned lds_off_of(const void* p) {
  return (unsigned)(uintptr_t)p;  // low 32 bits of LDS-aperture address
}

// A-fragment (16x32, 16-bit): lane (lo,g): halves 0..7 <-> K = 8g+j,
// halves 8..15 <-> K = 16+8g+(j-8). Two 16B loads from a row.
__device__ __forceinline__ bf16x16 frag_a(const __bf16* p, int g) {
  bf16x8 lo = *(const bf16x8*)(p + 8 * g);
  bf16x8 hi = *(const bf16x8*)(p + 16 + 8 * g);
  bf16x16 r;
#pragma unroll
  for (int i = 0; i < 8; ++i) { r[i] = lo[i]; r[i + 8] = hi[i]; }
  return r;
}

// B-fragment (32x16, 16-bit): lane (lo,g): halves 0..7 <-> K = 8g+j,
// halves 8..15 <-> K = 16+8g+(j-8), column N = lo. Source must be transposed (row = N, contiguous in K).
__device__ __forceinline__ bf16x16 frag_b(const __bf16* p, int g) {
  bf16x8 lo = *(const bf16x8*)(p + 8 * g);
  bf16x8 hi = *(const bf16x8*)(p + 16 + 8 * g);
  bf16x16 r;
#pragma unroll
  for (int i = 0; i < 8; ++i) { r[i] = lo[i]; r[i + 8] = hi[i]; }
  return r;
}

// ---------------------------------------------------------------------------
// GEMM: C(MxN) = A(MxK,bf16 row-major) * B(KxN) with B given transposed
// Bt(NxK, bf16). Workgroup 256 thr = 8 waves (2x4), wave tile 32x64.
// Double-buffered async global->LDS staging of the 64x32 A and 256x32 B tiles.
// ---------------------------------------------------------------------------
template <typename OutT>
__global__ __launch_bounds__(256) void gemm_bf16_kernel(
    const __bf16* __restrict__ A, const __bf16* __restrict__ Bt,
    OutT* __restrict__ C, int M, int N, int K) {
  constexpr int LDA = 40;  // 64-row A tile, padded (80B rows, bank-skewed)
  constexpr int LDB = 40;  // 256-row B tile, padded
  __shared__ alignas(16) __bf16 As[2][64 * LDA];
  __shared__ alignas(16) __bf16 Bs[2][256 * LDB];

  const int tid  = threadIdx.x;
  const int lane = tid & 31;
  const int lo   = lane & 15;
  const int g    = lane >> 4;
  const int w    = tid >> 5;
  const int wrow = w >> 2;
  const int wcol = w & 3;
  const int m_base = blockIdx.y * 64;
  const int n_base = blockIdx.x * 256;

  const unsigned a_lds0 = lds_off_of(&As[0][0]);
  const unsigned b_lds0 = lds_off_of(&Bs[0][0]);

  // per-thread staging chunks: A = 1x16B, B = 4x16B
  const int ar = tid >> 2, ac = (tid & 3) * 8;

  auto stage = [&](int p, int kb) {
    async_b128(a_lds0 + (unsigned)(p * 64 * LDA + ar * LDA + ac) * 2u,
               A + (size_t)(m_base + ar) * K + kb + ac);
#pragma unroll
    for (int it = 0; it < 4; ++it) {
      const int ch = tid + 256 * it;
      const int br = ch >> 2, bc = (ch & 3) * 8;
      async_b128(b_lds0 + (unsigned)(p * 256 * LDB + br * LDB + bc) * 2u,
                 Bt + (size_t)(n_base + br) * K + kb + bc);
    }
  };

  const f32x8 zero = {0.f, 0.f, 0.f, 0.f, 0.f, 0.f, 0.f, 0.f};
  f32x8 acc[2][4];
#pragma unroll
  for (int i = 0; i < 2; ++i)
#pragma unroll
    for (int j = 0; j < 4; ++j) acc[i][j] = zero;

  stage(0, 0);
  wait_async0();
  __syncthreads();

  for (int kb = 0; kb < K; kb += 32) {
    const int p = (kb >> 5) & 1;
    if (kb + 32 < K) stage(p ^ 1, kb + 32);  // prefetch next K-step

    bf16x16 af[2], bv[4];
#pragma unroll
    for (int i = 0; i < 2; ++i)
      af[i] = frag_a(&As[p][(wrow * 32 + 16 * i + lo) * LDA], g);
#pragma unroll
    for (int j = 0; j < 4; ++j)
      bv[j] = frag_b(&Bs[p][(wcol * 64 + 16 * j + lo) * LDB], g);
#pragma unroll
    for (int i = 0; i < 2; ++i)
#pragma unroll
      for (int j = 0; j < 4; ++j)
        acc[i][j] = WMMA_BF16(af[i], bv[j], acc[i][j]);

    wait_async0();     // own wave's async issues complete
    __syncthreads();   // all waves' stages visible; ds reads drained
  }

  const int m0 = m_base + wrow * 32;
  const int n0 = n_base + wcol * 64;
#pragma unroll
  for (int i = 0; i < 2; ++i)
#pragma unroll
    for (int j = 0; j < 4; ++j)
#pragma unroll
      for (int r = 0; r < 8; ++r)
        C[(size_t)(m0 + 16 * i + r + 8 * g) * N + (n0 + 16 * j + lo)] =
            (OutT)acc[i][j][r];
}

// ---------------------------------------------------------------------------
// Elementwise converts / transposes
// ---------------------------------------------------------------------------
__global__ void cvt_bf16_kernel(const float* __restrict__ in,
                                __bf16* __restrict__ out, int n) {
  int i = blockIdx.x * 256 + threadIdx.x;
  if (i < n) out[i] = (__bf16)in[i];
}

// in: fp32 (R x Cc) -> out: bf16 (Cc x R)
__global__ void transpose_cvt_kernel(const float* __restrict__ in,
                                     __bf16* __restrict__ out, int R, int Cc) {
  size_t i = (size_t)blockIdx.x * 256 + threadIdx.x;
  if (i < (size_t)R * Cc) {
    int r = (int)(i / Cc);
    int c = (int)(i % Cc);
    out[(size_t)c * R + r] = (__bf16)in[i];
  }
}

// V (B*T x C) bf16 -> Vt (B,NH,HD,T) bf16
__global__ void vtrans_kernel(const __bf16* __restrict__ V,
                              __bf16* __restrict__ Vt) {
  size_t i = (size_t)blockIdx.x * 256 + threadIdx.x;  // over 4096*1024
  int col = (int)(i & 1023);
  int row = (int)(i >> 10);
  int b = row >> 10, t = row & 1023;
  int h = col >> 6,  d = col & 63;
  Vt[((size_t)((b * NHn + h) * HDn + d) << 10) | (size_t)t] = V[i];
}

// ---------------------------------------------------------------------------
// lambda scalar: exp(sum lq1*lk1) - exp(sum lq2*lk2) + LAMBDA_INIT
// ---------------------------------------------------------------------------
__global__ void lam_kernel(const float* lq1, const float* lk1,
                           const float* lq2, const float* lk2, float* lam) {
  __shared__ float s1[64], s2[64];
  int t = threadIdx.x;
  s1[t] = lq1[t] * lk1[t];
  s2[t] = lq2[t] * lk2[t];
  __syncthreads();
  if (t == 0) {
    float a = 0.f, b = 0.f;
    for (int i = 0; i < 64; ++i) { a += s1[i]; b += s2[i]; }
    *lam = __expf(a) - __expf(b) + LAMBDA_INIT;
  }
}

// ---------------------------------------------------------------------------
// Differential attention core. One wave handles one 16-query tile of (b,h).
// Scores computed transposed (S^T = K * Q^T) so the C-layout of the score
// tile matches the A-fragment layout of the probability tile -> no cross-lane
// shuffles between softmax and the P*V WMMA.
// ---------------------------------------------------------------------------
__global__ __launch_bounds__(128) void diffattn_kernel(
    const __bf16* __restrict__ Q, const __bf16* __restrict__ Kmat,
    const __bf16* __restrict__ Vt, const float* __restrict__ lam_p,
    float* __restrict__ Yraw) {
  const int lane = threadIdx.x & 31;
  const int lo   = lane & 15;   // N index: query within tile (scores), d (acc)
  const int g    = lane >> 4;
  const int w    = threadIdx.x >> 5;
  const int gw   = blockIdx.x * 4 + w;        // 4096 query tiles total
  const int qt = gw & (Tn / 16 - 1);
  const int h  = (gw >> 6) & (NHn - 1);
  const int b  = gw >> 10;
  const int q0 = qt * 16;
  const float lam   = lam_p[0];
  const float scale = 0.125f;  // 1/sqrt(64)

  // Q/K layout: (B*T, 2C) with col = h*128 + split*64 + d
  const __bf16* qrow  = Q    + (size_t)(b * Tn + q0 + lo) * (2 * Cn) + h * 2 * HDn;
  const __bf16* Kbase = Kmat + (size_t)(b * Tn) * (2 * Cn) + h * 2 * HDn;
  const __bf16* Vth   = Vt   + (size_t)((b * NHn + h) * HDn) * Tn;

  // B-fragments of Q (column = query), two 32-wide d-steps each
  bf16x16 bq1[2], bq2[2];
#pragma unroll
  for (int s = 0; s < 2; ++s) {
    bq1[s] = frag_b(qrow + s * 32, g);
    bq2[s] = frag_b(qrow + HDn + s * 32, g);
  }

  const f32x8 zero = {0.f, 0.f, 0.f, 0.f, 0.f, 0.f, 0.f, 0.f};
  f32x8 acc1[4], acc2[4];
#pragma unroll
  for (int dt = 0; dt < 4; ++dt) { acc1[dt] = zero; acc2[dt] = zero; }
  float l1 = 0.f, l2 = 0.f;

  const int nch = (q0 + 47) / 32;  // 32-key chunks covering keys 0..q0+15
  for (int ci = 0; ci < nch; ++ci) {
    const int c = ci * 32;
    f32x8 s1t0 = zero, s1t1 = zero, s2t0 = zero, s2t1 = zero;
#pragma unroll
    for (int s = 0; s < 2; ++s) {
      bf16x16 a10 = frag_a(Kbase + (size_t)(c + lo) * (2 * Cn) + s * 32, g);
      bf16x16 a11 = frag_a(Kbase + (size_t)(c + 16 + lo) * (2 * Cn) + s * 32, g);
      bf16x16 a20 = frag_a(Kbase + (size_t)(c + lo) * (2 * Cn) + HDn + s * 32, g);
      bf16x16 a21 = frag_a(Kbase + (size_t)(c + 16 + lo) * (2 * Cn) + HDn + s * 32, g);
      s1t0 = WMMA_BF16(a10, bq1[s], s1t0);
      s1t1 = WMMA_BF16(a11, bq1[s], s1t1);
      s2t0 = WMMA_BF16(a20, bq2[s], s2t0);
      s2t1 = WMMA_BF16(a21, bq2[s], s2t1);
    }
    // exp + causal mask; C-layout reg r of half g holds key c+r+8g for query lo
    bf16x16 pa1, pa2;
    const int qg = q0 + lo;
#pragma unroll
    for (int r = 0; r < 8; ++r) {
      const int k0 = c + r + 8 * g;
      const int k1 = k0 + 16;
      float e;
      e = (k0 <= qg) ? __expf(s1t0[r] * scale) : 0.f; l1 += e; pa1[r]     = (__bf16)e;
      e = (k1 <= qg) ? __expf(s1t1[r] * scale) : 0.f; l1 += e; pa1[r + 8] = (__bf16)e;
      e = (k0 <= qg) ? __expf(s2t0[r] * scale) : 0.f; l2 += e; pa2[r]     = (__bf16)e;
      e = (k1 <= qg) ? __expf(s2t1[r] * scale) : 0.f; l2 += e; pa2[r + 8] = (__bf16)e;
    }
    // P(16q x 32k) * V(32k x 64d); Vt rows are d, contiguous in keys
#pragma unroll
    for (int dt = 0; dt < 4; ++dt) {
      bf16x16 bv = frag_b(Vth + (size_t)(dt * 16 + lo) * Tn + c, g);
      acc1[dt] = WMMA_BF16(pa1, bv, acc1[dt]);
      acc2[dt] = WMMA_BF16(pa2, bv, acc2[dt]);
    }
  }

  // combine the two key-half partial sums; l indexed by query = lo
  l1 += __shfl_xor(l1, 16, 32);
  l2 += __shfl_xor(l2, 16, 32);

  __shared__ float lsh[4][2][16];
  if (g == 0) { lsh[w][0][lo] = l1; lsh[w][1][lo] = l2; }
  __syncthreads();

  float inv1[8], inv2[8];
#pragma unroll
  for (int r = 0; r < 8; ++r) {
    inv1[r] = 1.0f / lsh[w][0][r + 8 * g];
    inv2[r] = lam / lsh[w][1][r + 8 * g];
  }

  // acc layout: lane lo = d within tile, reg r of half g = query r+8g
  float* yout = Yraw + (size_t)((b * NHn + h) * Tn + q0) * HDn;
#pragma unroll
  for (int dt = 0; dt < 4; ++dt)
#pragma unroll
    for (int r = 0; r < 8; ++r) {
      const int q = r + 8 * g;
      yout[(size_t)q * HDn + dt * 16 + lo] =
          acc1[dt][r] * inv1[r] - acc2[dt][r] * inv2[r];
    }
}

// ---------------------------------------------------------------------------
// GroupNorm stats per (b,h): mean and rsqrt(var+eps)*(1-lambda_init)
// ---------------------------------------------------------------------------
__global__ __launch_bounds__(256) void gn_stats_kernel(
    const float* __restrict__ Y, float* __restrict__ stats) {
  const int bh = blockIdx.x;
  const float* p = Y + (size_t)bh * Tn * HDn;
  float s = 0.f, ss = 0.f;
  for (int i = threadIdx.x; i < Tn * HDn; i += 256) {
    float v = p[i];
    s += v;
    ss += v * v;
  }
  __shared__ float sh1[256], sh2[256];
  sh1[threadIdx.x] = s;
  sh2[threadIdx.x] = ss;
  __syncthreads();
  for (int off = 128; off > 0; off >>= 1) {
    if (threadIdx.x < off) {
      sh1[threadIdx.x] += sh1[threadIdx.x + off];
      sh2[threadIdx.x] += sh2[threadIdx.x + off];
    }
    __syncthreads();
  }
  if (threadIdx.x == 0) {
    const float n = (float)(Tn * HDn);
    float mean = sh1[0] / n;
    float var  = sh2[0] / n - mean * mean;
    stats[bh * 2]     = mean;
    stats[bh * 2 + 1] = rsqrtf(var + 1e-5f) * ONE_MINUS_LI;
  }
}

// normalize + transpose (B,NH,T,HD) -> (B,T,C) bf16
__global__ void norm_kernel(const float* __restrict__ Y,
                            const float* __restrict__ stats,
                            __bf16* __restrict__ Yn) {
  size_t idx = (size_t)blockIdx.x * 256 + threadIdx.x;  // over B*NH*T*HD
  int d = (int)(idx & 63);
  int t = (int)((idx >> 6) & 1023);
  int h = (int)((idx >> 16) & 15);
  int b = (int)(idx >> 20);
  int bh = (b << 4) | h;
  float v = (Y[idx] - stats[bh * 2]) * stats[bh * 2 + 1];
  Yn[(size_t)(b * Tn + t) * Cn + h * HDn + d] = (__bf16)v;
}

// ---------------------------------------------------------------------------
// Host orchestration
// ---------------------------------------------------------------------------
extern "C" void kernel_launch(void* const* d_in, const int* in_sizes, int n_in,
                              void* d_out, int out_size, void* d_ws,
                              size_t ws_size, hipStream_t stream) {
  (void)in_sizes; (void)n_in; (void)out_size; (void)ws_size;
  const float* x   = (const float*)d_in[0];
  const float* Wq  = (const float*)d_in[1];
  const float* Wk  = (const float*)d_in[2];
  const float* Wv  = (const float*)d_in[3];
  const float* Wc  = (const float*)d_in[4];
  const float* lq1 = (const float*)d_in[5];
  const float* lk1 = (const float*)d_in[6];
  const float* lq2 = (const float*)d_in[7];
  const float* lk2 = (const float*)d_in[8];

  char* ws = (char*)d_ws;
  size_t off = 0;
  auto alloc = [&](size_t bytes) -> char* {
    char* p = ws + off;
    off = (off + bytes + 255) & ~(size_t)255;
    return p;
  };

  const size_t BT = (size_t)Bn * Tn;  // 4096
  __bf16* Xb   = (__bf16*)alloc(BT * Cn * 2);
  __bf16* Wqt  = (__bf16*)alloc((size_t)2 * Cn * Cn * 2);
  __bf16* Wkt  = (__bf16*)alloc((size_t)2 * Cn * Cn * 2);
  __bf16* Wvt  = (__bf16*)alloc((size_t)Cn * Cn * 2);
  __bf16* Wct  = (__bf16*)alloc((size_t)Cn * Cn * 2);
  __bf16* Qb   = (__bf16*)alloc(BT * 2 * Cn * 2);
  __bf16* Kb   = (__bf16*)alloc(BT * 2 * Cn * 2);
  __bf16* Vb   = (__bf16*)alloc(BT * Cn * 2);
  __bf16* Vtb  = (__bf16*)alloc(BT * Cn * 2);
  float*  Yraw = (float*)alloc(BT * Cn * 4);
  __bf16* Yn   = (__bf16*)alloc(BT * Cn * 2);
  float*  stats = (float*)alloc(Bn * NHn * 2 * 4);
  float*  lam   = (float*)alloc(4);

  lam_kernel<<<1, 64, 0, stream>>>(lq1, lk1, lq2, lk2, lam);
  cvt_bf16_kernel<<<(int)(BT * Cn / 256), 256, 0, stream>>>(x, Xb, (int)(BT * Cn));
  transpose_cvt_kernel<<<(Cn * 2 * Cn) / 256, 256, 0, stream>>>(Wq, Wqt, Cn, 2 * Cn);
  transpose_cvt_kernel<<<(Cn * 2 * Cn) / 256, 256, 0, stream>>>(Wk, Wkt, Cn, 2 * Cn);
  transpose_cvt_kernel<<<(Cn * Cn) / 256, 256, 0, stream>>>(Wv, Wvt, Cn, Cn);
  transpose_cvt_kernel<<<(Cn * Cn) / 256, 256, 0, stream>>>(Wc, Wct, Cn, Cn);

  dim3 blk(256);
  gemm_bf16_kernel<__bf16><<<dim3((2 * Cn) / 256, (int)(BT / 64)), blk, 0, stream>>>(
      Xb, Wqt, Qb, (int)BT, 2 * Cn, Cn);
  gemm_bf16_kernel<__bf16><<<dim3((2 * Cn) / 256, (int)(BT / 64)), blk, 0, stream>>>(
      Xb, Wkt, Kb, (int)BT, 2 * Cn, Cn);
  gemm_bf16_kernel<__bf16><<<dim3(Cn / 256, (int)(BT / 64)), blk, 0, stream>>>(
      Xb, Wvt, Vb, (int)BT, Cn, Cn);

  vtrans_kernel<<<(int)(BT * Cn / 256), 256, 0, stream>>>(Vb, Vtb);

  diffattn_kernel<<<(Bn * NHn * (Tn / 16)) / 4, 128, 0, stream>>>(Qb, Kb, Vtb, lam, Yraw);

  gn_stats_kernel<<<Bn * NHn, 256, 0, stream>>>(Yraw, stats);
  norm_kernel<<<(int)(BT * Cn / 256), 256, 0, stream>>>(Yraw, stats, Yn);

  gemm_bf16_kernel<float><<<dim3(Cn / 256, (int)(BT / 64)), blk, 0, stream>>>(
      Yn, Wct, (float*)d_out, (int)BT, Cn, Cn);
}
